// NAPPGINBase_67053029425790
// MI455X (gfx1250) — matrix operand order repaired and checked by reading, not compiled, not run on hardware
//
#include <hip/hip_runtime.h>
#include <hip/hip_bf16.h>
#include <cstdint>

// ---------------- problem constants ----------------
constexpr int BQ   = 64;      // graphs
constexpr int NN   = 512;     // nodes per graph
constexpr int EQ   = 64 * 512 * 8; // edges = 262144
constexpr int CC   = 16;      // clusters
constexpr int RTOT = BQ * NN; // 32768 rows

// ---------------- workspace layout (bytes) ----------------
constexpr size_t OFF_ADJI = 0;                      // int32 adjT  [64][512][512]
constexpr size_t OFF_DEGI = 67108864;               // int32 deg   [32768]
constexpr size_t OFF_ADJB = 67239936;               // bf16 adjT   [64][512][512]
constexpr size_t OFF_HT   = 100794368;              // bf16 hT     [512][32768]
constexpr size_t OFF_ZSUM = 134348800;              // bf16 zsum   [32768][512]
constexpr size_t OFF_Z1   = 167903232;              // bf16 z1     [32768][512]
constexpr size_t OFF_Z2   = 201457664;              // bf16 z2     [32768][512]
constexpr size_t OFF_W1T  = 235012096;              // bf16 w1T    [512][512]
constexpr size_t OFF_W2T  = 235536384;              // bf16 w2T    [512][512]
constexpr size_t OFF_ST   = 236060672;              // bf16 sT     [16][32768]
constexpr size_t OFF_SF   = 237109248;              // f32  s      [32768][16]
constexpr size_t OFF_Y    = 239206400;              // f32  y      [32768][16]
constexpr size_t OFF_DEGF = 241303552;              // f32  deg    [32768]

typedef __bf16 bf16;
typedef bf16  v16bf __attribute__((ext_vector_type(16)));
typedef bf16  v8bf  __attribute__((ext_vector_type(8)));
typedef float v8f   __attribute__((ext_vector_type(8)));

__device__ inline v8f v8f_zero() {
  v8f z;
#pragma unroll
  for (int i = 0; i < 8; ++i) z[i] = 0.f;
  return z;
}

// A fragment: 16x32 bf16, row-major source, interleaved K halves per ISA layout.
// lane<16 holds K {0..7,16..23}; lane>=16 holds K {8..15,24..31} of row m=lane&15.
__device__ inline v16bf load_frag_a(const bf16* base, size_t ld) {
  const int lane = threadIdx.x & 31;
  const int m    = lane & 15;
  const int sel  = lane >> 4;
  const bf16* row = base + (size_t)m * ld;
  v8bf lo = *(const v8bf*)(row + (sel ? 8 : 0));
  v8bf hi = *(const v8bf*)(row + (sel ? 24 : 16));
  v16bf a;
#pragma unroll
  for (int e = 0; e < 8; ++e) { a[e] = lo[e]; a[e + 8] = hi[e]; }
  return a;
}

// B fragment: 32x16 bf16, loaded from B^T (N x K row-major).
// lane<16 holds K {0..7,16..23}, lane>=16 holds K {8..15,24..31} of column n=lane&15.
__device__ inline v16bf load_frag_b(const bf16* baseT, size_t ld) {
  const int lane = threadIdx.x & 31;
  const int n    = lane & 15;
  const int sel  = lane >> 4;
  const bf16* row = baseT + (size_t)n * ld + sel * 8;
  v8bf lo = *(const v8bf*)(row);
  v8bf hi = *(const v8bf*)(row + 16);
  v16bf b;
#pragma unroll
  for (int e = 0; e < 8; ++e) { b[e] = lo[e]; b[e + 8] = hi[e]; }
  return b;
}

__device__ inline v8f wmma_bf16(v16bf a, v16bf b, v8f c) {
  return __builtin_amdgcn_wmma_f32_16x16x32_bf16(false, a, false, b, (short)0, c,
                                                 false, false);
}

// ---------------- init: zero adjT+deg ints, zero output scalar ----------------
__global__ void k_init(int* p, float* out, int n) {
  int i = blockIdx.x * 256 + threadIdx.x;
  if (i < n) p[i] = 0;
  if (i == 0) out[0] = 0.f;
}

// ---------------- weight transpose + bf16 convert ----------------
__global__ void k_wconv(const float* w1, const float* w2, bf16* w1T, bf16* w2T) {
  int i = blockIdx.x * 256 + threadIdx.x;
  if (i < 262144) {
    int n = i >> 9, k = i & 511;
    w1T[i] = (bf16)w1[(size_t)k * 512 + n];
  } else if (i < 524288) {
    int o = i - 262144;
    int n = o >> 9, k = o & 511;
    w2T[o] = (bf16)w2[(size_t)k * 512 + n];
  }
}

// ---------------- gather: hT[j][i] = emb[x[i]][j] (bf16), coalesced writes ----
__global__ void k_gather(const int* x, const float* emb, bf16* hT) {
  int idx = blockIdx.x * 256 + threadIdx.x;
  if (idx >= 512 * RTOT) return;
  int j = idx >> 15;          // feature
  int i = idx & 32767;        // node
  hT[idx] = (bf16)emb[(size_t)x[i] * 512 + j];
}

// ---------------- edges: adjT[g][dst][src]++ ; deg[src]++ ----------------
__global__ void k_edges(const int* ei, int* adjT, int* degi) {
  int e = blockIdx.x * 256 + threadIdx.x;
  if (e >= EQ) return;
  int src = ei[e];
  int dst = ei[EQ + e];
  int g = src >> 9;
  atomicAdd(&adjT[((size_t)g << 18) + ((size_t)(dst & 511) << 9) + (src & 511)], 1);
  atomicAdd(&degi[src], 1);
}

// ---------------- convert adjT int->bf16, deg int->f32 ----------------
__global__ void k_adjconv(const int* adjI, bf16* adjB, const int* degI, float* degF) {
  int i = blockIdx.x * 256 + threadIdx.x;
  if (i < 16777216) adjB[i] = (bf16)(float)adjI[i];
  else if (i < 16777216 + RTOT) degF[i - 16777216] = (float)degI[i - 16777216];
}

// ---------------- GEMM: zsum = h + adjT @ h  (per graph) ----------------
// Each wave: 32x64 output (2 M-subtiles x 4 N-subtiles = 8 WMMA accumulators).
__global__ __launch_bounds__(256) void k_gemm_agg(const bf16* adjT, const bf16* hT,
                                                  const int* x, const float* emb,
                                                  bf16* zsum) {
  const int wave = threadIdx.x >> 5;
  const int lane = threadIdx.x & 31;
  const int g    = blockIdx.x >> 4;            // 16 blocks per graph
  const int job  = (blockIdx.x & 15) * 8 + wave;  // 0..127
  const int wm = job >> 3;                     // 0..15 (32-row strip)
  const int wn = job & 7;                      // 0..7  (64-col strip)
  const int i0 = wm * 32, n0 = wn * 64;
  const int sel = lane >> 4, nn = lane & 15;

  v8f acc[2][4];
#pragma unroll
  for (int u = 0; u < 2; ++u)
#pragma unroll
    for (int t = 0; t < 4; ++t) {
      int j = n0 + t * 16 + nn;
#pragma unroll
      for (int r = 0; r < 8; ++r) {
        int row = g * 512 + i0 + u * 16 + r + sel * 8;
        acc[u][t][r] = emb[(size_t)x[row] * 512 + j];   // C init = h (fp32)
      }
    }
  const bf16* Abase = adjT + ((size_t)g << 18) + (size_t)i0 * 512;
  for (int ks = 0; ks < 512; ks += 32) {
    v16bf a0 = load_frag_a(Abase + ks, 512);
    v16bf a1 = load_frag_a(Abase + (size_t)16 * 512 + ks, 512);
#pragma unroll
    for (int t = 0; t < 4; ++t) {
      v16bf b = load_frag_b(hT + (size_t)(n0 + t * 16) * 32768 + (size_t)g * 512 + ks,
                            32768);
      acc[0][t] = wmma_bf16(a0, b, acc[0][t]);
      acc[1][t] = wmma_bf16(a1, b, acc[1][t]);
    }
  }
#pragma unroll
  for (int u = 0; u < 2; ++u)
#pragma unroll
    for (int t = 0; t < 4; ++t) {
      int j = n0 + t * 16 + nn;
#pragma unroll
      for (int r = 0; r < 8; ++r) {
        int row = g * 512 + i0 + u * 16 + r + sel * 8;
        zsum[(size_t)row * 512 + j] = (bf16)acc[u][t][r];
      }
    }
}

// ---------------- GEMM: z1 = ELU(BN(zsum @ w1 + b1)) ----------------
__global__ __launch_bounds__(256) void k_gemm_bn_elu(const bf16* A, const bf16* BT,
                                                     const float* bias,
                                                     const float* gamma,
                                                     const float* beta,
                                                     const float* mean,
                                                     const float* var, bf16* out) {
  const int wave = threadIdx.x >> 5;
  const int lane = threadIdx.x & 31;
  const int job  = blockIdx.x * 8 + wave;
  const int wm = job >> 3, wn = job & 7;
  const int i0 = wm * 32, n0 = wn * 64;
  const int sel = lane >> 4, nn = lane & 15;

  v8f acc[2][4];
#pragma unroll
  for (int u = 0; u < 2; ++u)
#pragma unroll
    for (int t = 0; t < 4; ++t) acc[u][t] = v8f_zero();

  const bf16* Abase = A + (size_t)i0 * 512;
  for (int ks = 0; ks < 512; ks += 32) {
    __builtin_prefetch(Abase + ks + 128, 0, 1);   // global_prefetch_b8
    v16bf a0 = load_frag_a(Abase + ks, 512);
    v16bf a1 = load_frag_a(Abase + (size_t)16 * 512 + ks, 512);
#pragma unroll
    for (int t = 0; t < 4; ++t) {
      v16bf b = load_frag_b(BT + (size_t)(n0 + t * 16) * 512 + ks, 512);
      acc[0][t] = wmma_bf16(a0, b, acc[0][t]);
      acc[1][t] = wmma_bf16(a1, b, acc[1][t]);
    }
  }
#pragma unroll
  for (int t = 0; t < 4; ++t) {
    int j = n0 + t * 16 + nn;
    float sc = gamma[j] * rsqrtf(var[j] + 1e-5f);
    float sh = beta[j] - mean[j] * sc;
    float bb = bias[j];
#pragma unroll
    for (int u = 0; u < 2; ++u)
#pragma unroll
      for (int r = 0; r < 8; ++r) {
        int row = i0 + u * 16 + r + sel * 8;
        float v = acc[u][t][r] + bb;
        v = v * sc + sh;
        v = v > 0.f ? v : (expf(v) - 1.f);
        out[(size_t)row * 512 + j] = (bf16)v;
      }
  }
}

// ---------------- GEMM: z2 = ELU(z1 @ w2 + b2) ----------------
__global__ __launch_bounds__(256) void k_gemm_elu(const bf16* A, const bf16* BT,
                                                  const float* bias, bf16* out) {
  const int wave = threadIdx.x >> 5;
  const int lane = threadIdx.x & 31;
  const int job  = blockIdx.x * 8 + wave;
  const int wm = job >> 3, wn = job & 7;
  const int i0 = wm * 32, n0 = wn * 64;
  const int sel = lane >> 4, nn = lane & 15;

  v8f acc[2][4];
#pragma unroll
  for (int u = 0; u < 2; ++u)
#pragma unroll
    for (int t = 0; t < 4; ++t) acc[u][t] = v8f_zero();

  const bf16* Abase = A + (size_t)i0 * 512;
  for (int ks = 0; ks < 512; ks += 32) {
    __builtin_prefetch(Abase + ks + 128, 0, 1);
    v16bf a0 = load_frag_a(Abase + ks, 512);
    v16bf a1 = load_frag_a(Abase + (size_t)16 * 512 + ks, 512);
#pragma unroll
    for (int t = 0; t < 4; ++t) {
      v16bf b = load_frag_b(BT + (size_t)(n0 + t * 16) * 512 + ks, 512);
      acc[0][t] = wmma_bf16(a0, b, acc[0][t]);
      acc[1][t] = wmma_bf16(a1, b, acc[1][t]);
    }
  }
#pragma unroll
  for (int t = 0; t < 4; ++t) {
    int j = n0 + t * 16 + nn;
    float bb = bias[j];
#pragma unroll
    for (int u = 0; u < 2; ++u)
#pragma unroll
      for (int r = 0; r < 8; ++r) {
        int row = i0 + u * 16 + r + sel * 8;
        float v = acc[u][t][r] + bb;
        v = v > 0.f ? v : (expf(v) - 1.f);
        out[(size_t)row * 512 + j] = (bf16)v;
      }
  }
}

// ---------------- pool logits + softmax -> s (f32) and sT (bf16) ----------------
__global__ __launch_bounds__(256) void k_pool(const bf16* z2, const float* pw,
                                              const float* pb, float* s, bf16* sT) {
  __shared__ float lpw[512 * 16];
  __shared__ float lpb[16];
  for (int i = threadIdx.x; i < 8192; i += 256) lpw[i] = pw[i];
  if (threadIdx.x < 16) lpb[threadIdx.x] = pb[threadIdx.x];
  __syncthreads();

  int row = blockIdx.x * 256 + threadIdx.x;
  float acc[16];
#pragma unroll
  for (int c = 0; c < 16; ++c) acc[c] = lpb[c];
  const bf16* zr = z2 + (size_t)row * 512;
  for (int k = 0; k < 512; ++k) {
    float zv = (float)zr[k];
#pragma unroll
    for (int c = 0; c < 16; ++c) acc[c] += zv * lpw[k * 16 + c];
  }
  float mx = acc[0];
#pragma unroll
  for (int c = 1; c < 16; ++c) mx = fmaxf(mx, acc[c]);
  float sum = 0.f;
#pragma unroll
  for (int c = 0; c < 16; ++c) { acc[c] = expf(acc[c] - mx); sum += acc[c]; }
  float inv = 1.f / sum;
#pragma unroll
  for (int c = 0; c < 16; ++c) {
    float sv = acc[c] * inv;
    s[(size_t)row * 16 + c]   = sv;
    sT[(size_t)c * 32768 + row] = (bf16)sv;
  }
}

// ---------------- GEMM: y = adjT @ s  (per graph, N=16) ----------------
__global__ __launch_bounds__(256) void k_gemm_y(const bf16* adjT, const bf16* sT,
                                                float* y) {
  const int wave = threadIdx.x >> 5;
  const int lane = threadIdx.x & 31;
  const int job  = blockIdx.x * 8 + wave;
  const int g  = job >> 5;
  const int wm = job & 31;
  const int i0 = wm * 16;
  const int sel = lane >> 4, nn = lane & 15;

  v8f acc = v8f_zero();
  const bf16* Abase = adjT + ((size_t)g << 18) + (size_t)i0 * 512;
  for (int ks = 0; ks < 512; ks += 32) {
    v16bf a = load_frag_a(Abase + ks, 512);
    v16bf b = load_frag_b(sT + (size_t)g * 512 + ks, 32768);
    acc = wmma_bf16(a, b, acc);
  }
#pragma unroll
  for (int r = 0; r < 8; ++r) {
    int row = g * 512 + i0 + r + sel * 8;
    y[(size_t)row * 16 + nn] = acc[r];
  }
}

// ---------------- per-graph loss reduction ----------------
__global__ __launch_bounds__(256) void k_reduce(const float* s, const float* deg,
                                                const float* y, float* out) {
  const int g = blockIdx.x;
  __shared__ float ls[512 * 16];
  __shared__ float ldv[512];
  __shared__ float ca[16], cs[16], ssm[256];
  __shared__ float trv, dsum;
  const int tid = threadIdx.x;
  const float* sg = s + (size_t)g * 512 * 16;
  const float* yg = y + (size_t)g * 512 * 16;
  const float* dg = deg + (size_t)g * 512;

  for (int i = tid; i < 8192; i += 256) ls[i] = sg[i];
  for (int i = tid; i < 512; i += 256) ldv[i] = dg[i];
  if (tid < 16) { ca[tid] = 0.f; cs[tid] = 0.f; }
  if (tid == 0) { trv = 0.f; dsum = 0.f; }
  __syncthreads();

  float pca[16], pcs[16];
#pragma unroll
  for (int c = 0; c < 16; ++c) { pca[c] = 0.f; pcs[c] = 0.f; }
  float ptr = 0.f, pd = 0.f;
  for (int i = tid; i < 512; i += 256) {
    float d = ldv[i];
    pd += d;
#pragma unroll
    for (int c = 0; c < 16; ++c) {
      float sv = ls[i * 16 + c];
      pca[c] += sv * d;
      pcs[c] += sv;
      ptr += sv * yg[i * 16 + c];
    }
  }
#pragma unroll
  for (int c = 0; c < 16; ++c) { atomicAdd(&ca[c], pca[c]); atomicAdd(&cs[c], pcs[c]); }
  atomicAdd(&trv, ptr);
  atomicAdd(&dsum, pd);

  // ss[a][b] = sum_i s[i][a]*s[i][b], one (a,b) pair per thread
  const int aa = tid >> 4, bb = tid & 15;
  float ssp = 0.f;
  for (int i = 0; i < 512; ++i) ssp += ls[i * 16 + aa] * ls[i * 16 + bb];
  ssm[tid] = ssp;
  __syncthreads();

  if (tid == 0) {
    float m = dsum * 0.5f;
    float can = 0.f;
#pragma unroll
    for (int c = 0; c < 16; ++c) can += ca[c] * ca[c];
    float spectral = -(trv - can / (2.f * m)) / (2.f * m);
    float fro2 = 0.f;
    for (int p = 0; p < 256; ++p) fro2 += ssm[p] * ssm[p];
    float fro = sqrtf(fro2);
    float o2 = 0.f;
    for (int p = 0; p < 256; ++p) {
      float id = ((p >> 4) == (p & 15)) ? 0.25f : 0.f;
      float dv = ssm[p] / fro - id;
      o2 += dv * dv;
    }
    float ortho = sqrtf(o2);
    float cl2 = 0.f;
#pragma unroll
    for (int c = 0; c < 16; ++c) cl2 += cs[c] * cs[c];
    float cluster = sqrtf(cl2) / 512.f * 4.f - 1.f;
    atomicAdd(out, (spectral + ortho + cluster) * (1.f / 64.f));
  }
}

extern "C" void kernel_launch(void* const* d_in, const int* in_sizes, int n_in,
                              void* d_out, int out_size, void* d_ws, size_t ws_size,
                              hipStream_t stream) {
  const int*   x     = (const int*)d_in[0];
  const int*   ei    = (const int*)d_in[1];
  const float* emb   = (const float*)d_in[3];
  const float* w1    = (const float*)d_in[4];
  const float* b1    = (const float*)d_in[5];
  const float* gamma = (const float*)d_in[6];
  const float* beta  = (const float*)d_in[7];
  const float* mean  = (const float*)d_in[8];
  const float* var   = (const float*)d_in[9];
  const float* w2    = (const float*)d_in[10];
  const float* b2    = (const float*)d_in[11];
  const float* pw    = (const float*)d_in[12];
  const float* pb    = (const float*)d_in[13];

  char* ws = (char*)d_ws;
  int*   adjI = (int*)(ws + OFF_ADJI);
  int*   degI = (int*)(ws + OFF_DEGI);
  bf16*  adjB = (bf16*)(ws + OFF_ADJB);
  bf16*  hT   = (bf16*)(ws + OFF_HT);
  bf16*  zsum = (bf16*)(ws + OFF_ZSUM);
  bf16*  z1   = (bf16*)(ws + OFF_Z1);
  bf16*  z2   = (bf16*)(ws + OFF_Z2);
  bf16*  w1T  = (bf16*)(ws + OFF_W1T);
  bf16*  w2T  = (bf16*)(ws + OFF_W2T);
  bf16*  sT   = (bf16*)(ws + OFF_ST);
  float* sF   = (float*)(ws + OFF_SF);
  float* yF   = (float*)(ws + OFF_Y);
  float* degF = (float*)(ws + OFF_DEGF);
  float* out  = (float*)d_out;

  const int nzero = 16777216 + RTOT;  // adjT ints + deg ints (contiguous)
  k_init<<<(nzero + 255) / 256, 256, 0, stream>>>(adjI, out, nzero);
  k_wconv<<<(524288 + 255) / 256, 256, 0, stream>>>(w1, w2, w1T, w2T);
  k_gather<<<(512 * RTOT + 255) / 256, 256, 0, stream>>>(x, emb, hT);
  k_edges<<<(EQ + 255) / 256, 256, 0, stream>>>(ei, adjI, degI);
  k_adjconv<<<(nzero + 255) / 256, 256, 0, stream>>>(adjI, adjB, degI, degF);
  k_gemm_agg<<<BQ * 16, 256, 0, stream>>>(adjB, hT, x, emb, zsum);      // 1024 blocks
  k_gemm_bn_elu<<<1024, 256, 0, stream>>>(zsum, w1T, b1, gamma, beta, mean, var, z1);
  k_gemm_elu<<<1024, 256, 0, stream>>>(z1, w2T, b2, z2);
  k_pool<<<RTOT / 256, 256, 0, stream>>>(z2, pw, pb, sF, sT);
  k_gemm_y<<<256, 256, 0, stream>>>(adjB, sT, yF);
  k_reduce<<<BQ, 256, 0, stream>>>(sF, degF, yF, out);
}
